// GCN_VAE_t_84662395339252
// MI455X (gfx1250) — compile-verified, matrix operand order checked by reading, not run on hardware
//
#include <hip/hip_runtime.h>
#include <hip/hip_bf16.h>
#include <math.h>

typedef __attribute__((ext_vector_type(16))) _Float16     v16h;
typedef __attribute__((ext_vector_type(8)))  float        v8f;
typedef __attribute__((ext_vector_type(4)))  unsigned int v4u;

#define ACT_NONE    0
#define ACT_LEAKY   1
#define ACT_RELU    2
#define ACT_SIGMOID 3

// pack two f32 -> packed f16x2 (lowers to v_cvt_pk_f16_f32)
__device__ __forceinline__ unsigned int pack_f16x2(float a, float b) {
    _Float16 ha = (_Float16)a, hb = (_Float16)b;
    unsigned short ua, ub;
    __builtin_memcpy(&ua, &ha, 2);
    __builtin_memcpy(&ub, &hb, 2);
    return (unsigned int)ua | ((unsigned int)ub << 16);
}

union frag16 { v16h v; v4u q[2]; };

// Y[nrows x N_OUT] = act( X[nrows x K_IN (row stride ldx, col offset col0)] * W^T + bias )
// W stored [N_OUT x K_IN] row-major, so B[k][n] = W[n][k].
//
// LDS staging layout per row r (A-row m or B-row n): KP/2 packed dwords, where
// packed dword j holds f16 elements (2j, 2j+1). Per the CDNA5 16-bit 16x32
// fragment layout, lane group g (lane>>4) k-step ks needs k-pair dwords
// [ks*16+g*4 .. +3] and [+8 .. +11] -> each fragment = two ds_load_b128.
//
// Epilogue: acc fragments -> padded LDS tile -> flat coalesced stores.
// Requires 16*N_OUT % 32 == 0 (true for all layers here) for a fully
// predication-free store loop.
template <int K_IN, int N_OUT, int ACT>
__global__ __launch_bounds__(256) void wmma_layer_kernel(
    const float* __restrict__ X, int ldx, int col0,
    const float* __restrict__ W, const float* __restrict__ bias,
    float* __restrict__ Y, int nrows)
{
    constexpr int KSTEPS = (K_IN + 31) / 32;
    constexpr int NTILES = (N_OUT + 15) / 16;
    constexpr int KP  = KSTEPS * 32;   // padded K
    constexpr int NP  = NTILES * 16;   // padded N
    constexpr int KPW = KP / 2;        // packed dwords per row
    static_assert((16 * N_OUT) % 32 == 0, "store loop must be divergence-free");

    __shared__ __align__(16) unsigned int sW[NP * KPW];        // <= 8 KB
    __shared__ __align__(16) unsigned int sA[8 * 16 * KPW];    // <= 16 KB
    __shared__ __align__(16) float        sOut[8 * 16 * NP];   // <= 32 KB

    const int lane  = threadIdx.x & 31;
    const int wv    = threadIdx.x >> 5;                 // wave in block (0..7)
    const int wave  = (blockIdx.x * blockDim.x + threadIdx.x) >> 5;
    const int m0    = wave * 16;
    const int group = lane >> 4;
    const int ln    = lane & 15;

    // ---- cooperative W staging: f32 -> padded f16 image (branchless clamp+mask)
    for (int idx = threadIdx.x; idx < NP * KPW; idx += 256) {
        const int n  = idx / KPW;
        const int kp = (idx - n * KPW) * 2;
        const int nc  = (n < N_OUT) ? n : 0;
        const int kc0 = (kp     < K_IN) ? kp     : 0;
        const int kc1 = (kp + 1 < K_IN) ? kp + 1 : 0;
        const float* __restrict__ wrow = W + (size_t)nc * K_IN;
        float f0 = wrow[kc0];
        float f1 = wrow[kc1];
        f0 = (n < N_OUT && kp     < K_IN) ? f0 : 0.0f;
        f1 = (n < N_OUT && kp + 1 < K_IN) ? f1 : 0.0f;
        sW[idx] = pack_f16x2(f0, f1);
    }
    __syncthreads();   // all threads reach this; no earlier returns

    if (m0 >= nrows) return;   // wave-uniform -> EXEC all-ones below

    // ---- per-wave A staging (no barrier needed: same-wave LDS ordering)
    unsigned int* __restrict__ sAw = sA + wv * 16 * KPW;
    for (int idx = lane; idx < 16 * KPW; idx += 32) {
        const int r  = idx / KPW;
        const int kp = (idx - r * KPW) * 2;
        const int m  = m0 + r;
        const int mc  = (m < nrows) ? m : (nrows - 1);
        const int kc0 = (kp     < K_IN) ? kp     : 0;
        const int kc1 = (kp + 1 < K_IN) ? kp + 1 : 0;
        const float* __restrict__ xrow = X + (size_t)mc * ldx + col0;
        float f0 = xrow[kc0];
        float f1 = xrow[kc1];
        f0 = (m < nrows && kp     < K_IN) ? f0 : 0.0f;
        f1 = (m < nrows && kp + 1 < K_IN) ? f1 : 0.0f;
        sAw[idx] = pack_f16x2(f0, f1);
    }

    // ---- main WMMA loop: pure ds_load_b128 + v_wmma
    v8f acc[NTILES];
#pragma unroll
    for (int t = 0; t < NTILES; ++t) acc[t] = (v8f){0.f,0.f,0.f,0.f,0.f,0.f,0.f,0.f};

#pragma unroll
    for (int ks = 0; ks < KSTEPS; ++ks) {
        frag16 A;
        const int abase = ln * KPW + ks * 16 + group * 4;
        A.q[0] = *(const v4u*)&sAw[abase];
        A.q[1] = *(const v4u*)&sAw[abase + 8];
#pragma unroll
        for (int t = 0; t < NTILES; ++t) {
            frag16 B;
            const int bbase = (t * 16 + ln) * KPW + ks * 16 + group * 4;
            B.q[0] = *(const v4u*)&sW[bbase];
            B.q[1] = *(const v4u*)&sW[bbase + 8];
            acc[t] = __builtin_amdgcn_wmma_f32_16x16x32_f16(
                false, A.v, false, B.v, (short)0, acc[t], false, false);
        }
    }

    // ---- epilogue: acc -> padded LDS tile (unconditional ds stores)
    float* __restrict__ sOw = sOut + wv * 16 * NP;
#pragma unroll
    for (int t = 0; t < NTILES; ++t) {
#pragma unroll
        for (int v = 0; v < 8; ++v) {
            // C/D layout: VGPR v, lane -> row (v + 8*group), col (t*16 + ln)
            sOw[(v + 8 * group) * NP + t * 16 + ln] = acc[t][v];
        }
    }

    // ---- flat, predication-free, coalesced store of 16 x N_OUT tile
    float* __restrict__ yrow = Y + (size_t)m0 * N_OUT;
    for (int idx = lane; idx < 16 * N_OUT; idx += 32) {
        const int r = idx / N_OUT;
        const int c = idx - r * N_OUT;
        float val = sOw[r * NP + c];
        if (bias) val += bias[c];
        if (ACT == ACT_LEAKY)        val = (val >= 0.0f) ? val : 0.01f * val;
        else if (ACT == ACT_RELU)    val = fmaxf(val, 0.0f);
        else if (ACT == ACT_SIGMOID) val = 1.0f / (1.0f + __expf(-val));
        yrow[idx] = val;   // idx == r*N_OUT + c
    }
}

// ---------------- sparse / pointwise helpers ----------------

__global__ void set_ones_f32(float* __restrict__ p, int n) {
    int i = blockIdx.x * blockDim.x + threadIdx.x;
    if (i < n) p[i] = 1.0f;                       // self-loop contributes 1 to deg
}

__global__ void count_deg(const int* __restrict__ dst, float* __restrict__ deg, int e) {
    int i = blockIdx.x * blockDim.x + threadIdx.x;
    if (i < e) atomicAdd(&deg[dst[i]], 1.0f);
}

__global__ void rsqrt_deg(const float* __restrict__ deg, float* __restrict__ dinv, int n) {
    int i = blockIdx.x * blockDim.x + threadIdx.x;
    if (i < n) dinv[i] = rsqrtf(deg[i]);          // deg >= 1 always (self loop)
}

// agg = bg + hh * dinv^2  (self-loop term + bias), flat over n*50
__global__ void self_loop_bias(const float* __restrict__ hh, const float* __restrict__ dinv,
                               const float* __restrict__ bg, float* __restrict__ agg, int n) {
    int i = blockIdx.x * blockDim.x + threadIdx.x;
    if (i < n * 50) {
        int node = i / 50, c = i - node * 50;
        float di = dinv[node];
        agg[i] = bg[c] + hh[i] * di * di;
    }
}

// agg[dst] += hh[src] * dinv[src]*dinv[dst]  (hh/agg are 40 MB each -> L2 resident)
__global__ void scatter_edges(const int* __restrict__ src, const int* __restrict__ dst,
                              const float* __restrict__ hh, const float* __restrict__ dinv,
                              float* __restrict__ agg, int e) {
    int i = blockIdx.x * blockDim.x + threadIdx.x;
    if (i < e) {
        int s = src[i], d = dst[i];
        float w = dinv[s] * dinv[d];
        const float* __restrict__ hs = hh + (size_t)s * 50;
        float* __restrict__ ad = agg + (size_t)d * 50;
#pragma unroll
        for (int c = 0; c < 50; ++c) atomicAdd(&ad[c], hs[c] * w);
    }
}

// split e3 -> mu, log_var; z1 = mu + eps*exp(0.5*log_var)
__global__ void reparam_k(const float* __restrict__ e3, const float* __restrict__ eps,
                          float* __restrict__ mu, float* __restrict__ lv,
                          float* __restrict__ z1, int n) {
    int i = blockIdx.x * blockDim.x + threadIdx.x;
    if (i < n * 10) {
        int node = i / 10, c = i - node * 10;
        float m = e3[(size_t)node * 20 + c];
        float l = e3[(size_t)node * 20 + 10 + c];
        mu[i] = m;
        lv[i] = l;
        z1[i] = m + eps[i] * __expf(0.5f * l);
    }
}

// ---------------- launch ----------------

static inline int cdiv(int a, int b) { return (a + b - 1) / b; }

template <int K_IN, int N_OUT, int ACT>
static void launch_layer(const float* X, int ldx, int col0, const float* W,
                         const float* bias, float* Y, int nrows, hipStream_t stream) {
    int waves = cdiv(nrows, 16);
    dim3 grid(cdiv(waves, 8)), block(256);
    wmma_layer_kernel<K_IN, N_OUT, ACT><<<grid, block, 0, stream>>>(X, ldx, col0, W, bias, Y, nrows);
}

extern "C" void kernel_launch(void* const* d_in, const int* in_sizes, int n_in,
                              void* d_out, int out_size, void* d_ws, size_t ws_size,
                              hipStream_t stream) {
    const float* x   = (const float*)d_in[0];
    const int*   ei  = (const int*)d_in[1];
    const float* eps = (const float*)d_in[2];
    const float* Wg  = (const float*)d_in[3];
    const float* bg  = (const float*)d_in[4];
    const float* We1 = (const float*)d_in[5];
    const float* be1 = (const float*)d_in[6];
    const float* We2 = (const float*)d_in[7];
    const float* be2 = (const float*)d_in[8];
    const float* We3 = (const float*)d_in[9];
    const float* be3 = (const float*)d_in[10];
    const float* Wfc = (const float*)d_in[11];
    const float* bfc = (const float*)d_in[12];
    const float* Wd1 = (const float*)d_in[13];
    const float* bd1 = (const float*)d_in[14];
    const float* Wd2 = (const float*)d_in[15];
    const float* bd2 = (const float*)d_in[16];
    const float* Wd3 = (const float*)d_in[17];
    const float* bd3 = (const float*)d_in[18];

    const int N = in_sizes[0] / 150;
    const int E = in_sizes[1] / 2;
    const int* src = ei;
    const int* dst = ei + E;

    // workspace layout (floats), with buffer reuse: total N*132 floats (~106 MB)
    float* ws   = (float*)d_ws;
    float* bufA = ws;                      // N*50 : hh -> e1 -> z1 -> d2
    float* bufB = ws + (size_t)N * 50;     // N*50 : agg(h) -> e3
    float* bufC = ws + (size_t)N * 100;    // N*30 : e2 -> d1
    float* deg  = ws + (size_t)N * 130;    // N
    float* dinv = ws + (size_t)N * 131;    // N

    float* out      = (float*)d_out;       // [mu'(N*50) | mu(N*10) | logvar(N*10) | z(N*10)]
    float* mu_prime = out;
    float* mu_out   = out + (size_t)N * 50;
    float* lv_out   = out + (size_t)N * 60;
    float* z_out    = out + (size_t)N * 70;

    const int T = 256;

    // 1) degrees & normalization
    set_ones_f32<<<cdiv(N, T), T, 0, stream>>>(deg, N);
    count_deg<<<cdiv(E, T), T, 0, stream>>>(dst, deg, E);
    rsqrt_deg<<<cdiv(N, T), T, 0, stream>>>(deg, dinv, N);

    // 2) hh = x[:,100:150] @ Wg^T   (WMMA)
    launch_layer<50, 50, ACT_NONE>(x, 150, 100, Wg, nullptr, bufA, N, stream);

    // 3) aggregate: self-loop + bias, then edge scatter (L2-resident)
    self_loop_bias<<<cdiv(N * 50, T), T, 0, stream>>>(bufA, dinv, bg, bufB, N);
    scatter_edges<<<cdiv(E, T), T, 0, stream>>>(src, dst, bufA, dinv, bufB, E);

    // 4) encoder (WMMA)
    launch_layer<50, 40, ACT_LEAKY>(bufB, 50, 0, We1, be1, bufA, N, stream);
    launch_layer<40, 30, ACT_LEAKY>(bufA, 40, 0, We2, be2, bufC, N, stream);
    launch_layer<30, 20, ACT_NONE >(bufC, 30, 0, We3, be3, bufB, N, stream);

    // 5) reparameterization -> mu, log_var outputs + z1 in bufA
    reparam_k<<<cdiv(N * 10, T), T, 0, stream>>>(bufB, eps, mu_out, lv_out, bufA, N);

    // 6) z = relu(z1 @ Wfc^T + bfc) -> output (also decoder input)
    launch_layer<10, 10, ACT_RELU>(bufA, 10, 0, Wfc, bfc, z_out, N, stream);

    // 7) decoder (WMMA)
    launch_layer<10, 20, ACT_LEAKY  >(z_out, 10, 0, Wd1, bd1, bufC, N, stream);
    launch_layer<20, 40, ACT_LEAKY  >(bufC, 20, 0, Wd2, bd2, bufA, N, stream);
    launch_layer<40, 50, ACT_SIGMOID>(bufA, 40, 0, Wd3, bd3, mu_prime, N, stream);

    (void)n_in; (void)out_size; (void)ws_size;
}
